// ConformerMoEEncoderLayer_29841432773032
// MI455X (gfx1250) — matrix operand order repaired and checked by reading, not compiled, not run on hardware
//
#include <hip/hip_runtime.h>
#include <hip/hip_bf16.h>

// ---------------------------------------------------------------------------
// Conformer-MoE encoder layer for MI455X (gfx1250, wave32, WMMA).
// All GEMMs: bf16 inputs, f32 accumulate via v_wmma_f32_16x16x32_bf16.
// Attention: flash-style (online softmax), never materializes [B,NH,T,T].
// GEMM tile staging uses GLOBAL_LOAD_ASYNC_TO_LDS_B128 (ASYNCcnt) when the
// toolchain exposes the builtin; falls back to load+ds_store otherwise.
// ---------------------------------------------------------------------------

#define BB 8
#define TT 1024
#define DD 512
#define HH 2048
#define NHH 8
#define DK 64
#define EE 8
#define KW 31
#define BT (BB * TT)   // 8192

typedef __attribute__((ext_vector_type(16))) __bf16 v16bf;
typedef __attribute__((ext_vector_type(8)))  float  v8f;
typedef __attribute__((ext_vector_type(4)))  int    v4i;

#if defined(__HIP_DEVICE_COMPILE__) && __has_builtin(__builtin_amdgcn_global_load_async_to_lds_b128)
#define USE_ASYNC_LDS 1
#else
#define USE_ASYNC_LDS 0
#endif

#if USE_ASYNC_LDS
typedef __attribute__((address_space(1))) v4i* as1_v4i_p;  // global
typedef __attribute__((address_space(3))) v4i* as3_v4i_p;  // LDS

__device__ __forceinline__ void async_g2l_b128(const unsigned short* g,
                                               unsigned short* l) {
  __builtin_amdgcn_global_load_async_to_lds_b128(
      (as1_v4i_p)(g), (as3_v4i_p)(l), 0, 0);
}
__device__ __forceinline__ void wait_async0() {
#if __has_builtin(__builtin_amdgcn_s_wait_asynccnt)
  __builtin_amdgcn_s_wait_asynccnt(0);
#else
  asm volatile("s_wait_asynccnt 0x0" ::: "memory");
#endif
}
#endif

struct Bf16Frag { uint4 lo, hi; };

__device__ __forceinline__ v16bf make_frag(uint4 lo, uint4 hi) {
  Bf16Frag f{lo, hi};
  return __builtin_bit_cast(v16bf, f);
}

__device__ __forceinline__ unsigned short f32_to_bf16(float f) {
  unsigned int u = __builtin_bit_cast(unsigned int, f);
  unsigned int r = (u + 0x7FFFu + ((u >> 16) & 1u)) >> 16;
  return (unsigned short)r;
}

__device__ __forceinline__ v8f wmma_bf16(v16bf a, v16bf b, v8f c) {
  return __builtin_amdgcn_wmma_f32_16x16x32_bf16(false, a, false, b, (short)0, c,
                                                 false, false);
}

// ---------------------------------------------------------------------------
// f32 -> bf16 convert (weights / activation restage)
// ---------------------------------------------------------------------------
__global__ void cvt_bf16_kernel(const float* __restrict__ in,
                                unsigned short* __restrict__ out, long n) {
  long i = (long)blockIdx.x * 256 + threadIdx.x;
  if (i < n) out[i] = f32_to_bf16(in[i]);
}

// ---------------------------------------------------------------------------
// chosen[b] = clip(dialectid[b]-1, 0, E-1)
// ---------------------------------------------------------------------------
__global__ void chosen_kernel(const int* __restrict__ did, int* __restrict__ ch) {
  int i = threadIdx.x;
  if (i < BB) {
    int c = did[i] - 1;
    c = c < 0 ? 0 : (c > (EE - 1) ? (EE - 1) : c);
    ch[i] = c;
  }
}

// ---------------------------------------------------------------------------
// LayerNorm over D=512; one row per block (256 threads, 2 elems/thread).
// Writes bf16 (GEMM staging) and/or f32 (final output).
// ---------------------------------------------------------------------------
__global__ __launch_bounds__(256) void ln_kernel(
    const float* __restrict__ x, const float* __restrict__ g,
    const float* __restrict__ b, unsigned short* __restrict__ out_bf,
    float* __restrict__ out_f32) {
  long row = blockIdx.x;
  const float* xr = x + row * DD;
  float v0 = xr[threadIdx.x];
  float v1 = xr[threadIdx.x + 256];
  float s = v0 + v1, sq = v0 * v0 + v1 * v1;
  #pragma unroll
  for (int off = 16; off > 0; off >>= 1) {
    s  += __shfl_xor(s,  off, 32);
    sq += __shfl_xor(sq, off, 32);
  }
  __shared__ float rs[8], rq[8], mv[2];
  int wid = threadIdx.x >> 5, lane = threadIdx.x & 31;
  if (lane == 0) { rs[wid] = s; rq[wid] = sq; }
  __syncthreads();
  if (threadIdx.x == 0) {
    float ts = 0.f, tq = 0.f;
    #pragma unroll
    for (int i = 0; i < 8; ++i) { ts += rs[i]; tq += rq[i]; }
    float mean = ts * (1.0f / DD);
    float var = tq * (1.0f / DD) - mean * mean;
    mv[0] = mean;
    mv[1] = rsqrtf(var + 1e-12f);
  }
  __syncthreads();
  float mean = mv[0], inv = mv[1];
  int i0 = threadIdx.x, i1 = threadIdx.x + 256;
  float y0 = (v0 - mean) * inv * g[i0] + b[i0];
  float y1 = (v1 - mean) * inv * g[i1] + b[i1];
  if (out_bf)  { out_bf[row * DD + i0] = f32_to_bf16(y0);
                 out_bf[row * DD + i1] = f32_to_bf16(y1); }
  if (out_f32) { out_f32[row * DD + i0] = y0;
                 out_f32[row * DD + i1] = y1; }
}

// ---------------------------------------------------------------------------
// bf16 WMMA GEMM: C[M,N] = act(A[M,K] @ B[K,N] + bias) (+ residual path)
// Block: 256 thr (8 waves), tile 128(M) x 64(N), BK=32.
// Wave grid 4x2; each wave: 2x2 WMMA accumulators (32x32).
// expert_sel != null: blockIdx.z = batch; B/bias indexed by chosen expert.
// mode: 0 = none, 1 = ReLU. residual != null: out = res + resScale * act(v).
// ---------------------------------------------------------------------------
__global__ __launch_bounds__(256) void gemm_bf16_kernel(
    const unsigned short* __restrict__ A, const unsigned short* __restrict__ Bw,
    const float* __restrict__ bias, const float* __restrict__ residual,
    float* __restrict__ Cf, unsigned short* __restrict__ Cbf,
    const int* __restrict__ expert_sel, int K, int N, int rowsPerBatch,
    int mode, float resScale) {
  const int batch = blockIdx.z;
  const long aRow0 = (long)batch * rowsPerBatch + (long)blockIdx.y * 128;
  const unsigned short* Bp = Bw;
  const float* biasP = bias;
  if (expert_sel) {
    long e = expert_sel[batch];
    Bp += e * (long)K * N;
    biasP += e * (long)N;
  }
  const int n0 = blockIdx.x * 64;

  __shared__ __align__(16) unsigned short sA[128 * 40];  // pad: 32 -> 40
  __shared__ __align__(16) unsigned short sB[32 * 72];   // pad: 64 -> 72

  const int wid = threadIdx.x >> 5, lane = threadIdx.x & 31;
  const int wm = wid >> 1, wn = wid & 1;
  const int r0 = (lane < 16) ? 0 : 8;
  const int m15 = lane & 15;

  v8f acc[2][2];
  acc[0][0] = v8f{}; acc[0][1] = v8f{}; acc[1][0] = v8f{}; acc[1][1] = v8f{};

  const int ar = threadIdx.x >> 1;          // 0..127
  const int ac = (threadIdx.x & 1) * 16;    // 0 / 16
  const int br = threadIdx.x >> 3;          // 0..31
  const int bc = (threadIdx.x & 7) * 8;     // 0..56

  const unsigned short* ga = A + (aRow0 + ar) * K + ac;
  const unsigned short* gb = Bp + (long)br * N + n0 + bc;

  for (int kc = 0; kc < K; kc += 32) {
#if USE_ASYNC_LDS
    // Direct global->LDS DMA copies (ASYNCcnt); no VGPR round-trip.
    async_g2l_b128(ga,     sA + ar * 40 + ac);
    async_g2l_b128(ga + 8, sA + ar * 40 + ac + 8);
    async_g2l_b128(gb,     sB + br * 72 + bc);
    if (kc + 32 < K) {
      __builtin_prefetch(ga + 32, 0, 3);
      __builtin_prefetch(gb + (long)32 * N, 0, 3);
    }
    wait_async0();        // this wave's copies landed in LDS
    __syncthreads();      // all waves' copies visible
#else
    *(uint4*)(sA + ar * 40 + ac)     = *(const uint4*)(ga);
    *(uint4*)(sA + ar * 40 + ac + 8) = *(const uint4*)(ga + 8);
    *(uint4*)(sB + br * 72 + bc)     = *(const uint4*)(gb);
    if (kc + 32 < K) {                     // global_prefetch for next k-chunk
      __builtin_prefetch(ga + 32, 0, 3);
      __builtin_prefetch(gb + (long)32 * N, 0, 3);
    }
    __syncthreads();
#endif

    v16bf af[2], bf[2];
    #pragma unroll
    for (int i = 0; i < 2; ++i) {
      const unsigned short* pa = sA + (wm * 32 + i * 16 + m15) * 40;
      af[i] = make_frag(*(const uint4*)(pa + r0), *(const uint4*)(pa + r0 + 16));
    }
    #pragma unroll
    for (int j = 0; j < 2; ++j) {
      const unsigned short* pb = sB + wn * 32 + j * 16 + m15;
      Bf16Frag fb;
      unsigned short* tb = (unsigned short*)&fb;
      #pragma unroll
      for (int s = 0; s < 8; ++s) { tb[s] = pb[(r0 + s) * 72]; tb[s + 8] = pb[(r0 + 16 + s) * 72]; }
      bf[j] = make_frag(fb.lo, fb.hi);
    }
    #pragma unroll
    for (int i = 0; i < 2; ++i)
      #pragma unroll
      for (int j = 0; j < 2; ++j)
        acc[i][j] = wmma_bf16(af[i], bf[j], acc[i][j]);
    __syncthreads();
    ga += 32;
    gb += (long)32 * N;
  }

  const int hi8 = (lane >> 4) * 8;
  #pragma unroll
  for (int i = 0; i < 2; ++i) {
    #pragma unroll
    for (int j = 0; j < 2; ++j) {
      #pragma unroll
      for (int r = 0; r < 8; ++r) {
        long row = aRow0 + wm * 32 + i * 16 + r + hi8;
        int col = n0 + wn * 32 + j * 16 + m15;
        float v = acc[i][j][r] + biasP[col];
        if (mode == 1) v = fmaxf(v, 0.f);
        long idx = row * N + col;
        float outv = v;
        if (residual) outv = residual[idx] + resScale * v;
        if (Cf)  Cf[idx] = outv;
        if (Cbf) Cbf[idx] = f32_to_bf16(outv);
      }
    }
  }
}

// ---------------------------------------------------------------------------
// Pack q/k/v f32 [B,T,NH*dk] -> bf16 [B,NH,T,dk]
// ---------------------------------------------------------------------------
__global__ void pack_qkv_kernel(const float* __restrict__ q,
                                const float* __restrict__ k,
                                const float* __restrict__ v,
                                unsigned short* __restrict__ qo,
                                unsigned short* __restrict__ ko,
                                unsigned short* __restrict__ vo) {
  long id = (long)blockIdx.x * 256 + threadIdx.x;  // < BT*D
  int c = (int)(id % DD);
  long bt = id / DD;
  int t = (int)(bt % TT);
  int b = (int)(bt / TT);
  int h = c >> 6, n = c & 63;
  long dst = (((long)(b * NHH + h)) * TT + t) * DK + n;
  qo[dst] = f32_to_bf16(q[id]);
  ko[dst] = f32_to_bf16(k[id]);
  vo[dst] = f32_to_bf16(v[id]);
}

// ---------------------------------------------------------------------------
// Flash attention: grid (T/128, NH, B); 8 waves, 16 query rows per wave.
// S = Q K^T via WMMA (K^T staged transposed in LDS), online softmax,
// P restaged through per-wave LDS as bf16 A-frag, O += P V via WMMA
// (V B-fragments loaded straight from global: dk column == lane, kv row == element).
// ---------------------------------------------------------------------------
__global__ __launch_bounds__(256) void flash_attn_kernel(
    const unsigned short* __restrict__ qbf, const unsigned short* __restrict__ kbf,
    const unsigned short* __restrict__ vbf, const unsigned char* __restrict__ mask,
    float* __restrict__ o) {
  const int b = blockIdx.z, h = blockIdx.y;
  const int w = threadIdx.x >> 5, lane = threadIdx.x & 31;
  const int q0 = blockIdx.x * 128 + w * 16;
  const long bh = (long)(b * NHH + h);
  const unsigned short* qp = qbf + bh * TT * DK;
  const unsigned short* kp = kbf + bh * TT * DK;
  const unsigned short* vp = vbf + bh * TT * DK;

  __shared__ __align__(16) unsigned short lds_kt[DK * 40];      // [dk][kv], pad 40
  __shared__ __align__(16) unsigned short lds_p[8 * 16 * 40];   // per-wave P

  const int m15 = lane & 15;
  const int r0 = (lane < 16) ? 0 : 8;
  const int hi8 = (lane >> 4) * 8;

  v16bf aq[2];
  #pragma unroll
  for (int c = 0; c < 2; ++c) {
    const unsigned short* base = qp + (long)(q0 + m15) * DK + c * 32;
    aq[c] = make_frag(*(const uint4*)(base + r0), *(const uint4*)(base + r0 + 16));
  }

  float mI[8], lI[8];
  v8f oacc[4];
  #pragma unroll
  for (int r = 0; r < 8; ++r) { mI[r] = -1e30f; lI[r] = 0.f; }
  #pragma unroll
  for (int t = 0; t < 4; ++t) oacc[t] = v8f{};

  unsigned short* ldsp = lds_p + w * 16 * 40;

  for (int kk = 0; kk < TT; kk += 32) {
    {  // stage K^T (transpose 32 x 64 -> [dk][kv])
      int row = threadIdx.x >> 3;           // kv 0..31
      int c0 = (threadIdx.x & 7) * 8;       // dk start
      uint4 vv = *(const uint4*)(kp + (long)(kk + row) * DK + c0);
      unsigned short tmp[8];
      *(uint4*)tmp = vv;
      #pragma unroll
      for (int j = 0; j < 8; ++j) lds_kt[(c0 + j) * 40 + row] = tmp[j];
    }
    __syncthreads();

    float sv[2][8];
    #pragma unroll
    for (int j = 0; j < 2; ++j) {
      v8f sacc = v8f{};
      #pragma unroll
      for (int c = 0; c < 2; ++c) {
        const unsigned short* pb = kp + (long)(kk + j * 16 + m15) * DK + c * 32;
        v16bf bk = make_frag(*(const uint4*)(pb + r0), *(const uint4*)(pb + r0 + 16));
        sacc = wmma_bf16(aq[c], bk, sacc);
      }
      int col = kk + j * 16 + m15;
      float mval = mask[b * TT + col] ? 0.f : -1e9f;
      #pragma unroll
      for (int r = 0; r < 8; ++r) sv[j][r] = sacc[r] * 0.125f + mval;
    }

    #pragma unroll
    for (int r = 0; r < 8; ++r) {
      float nm = fmaxf(mI[r], fmaxf(sv[0][r], sv[1][r]));
      #pragma unroll
      for (int off = 1; off < 16; off <<= 1) nm = fmaxf(nm, __shfl_xor(nm, off, 32));
      float alpha = __expf(mI[r] - nm);
      float p0 = __expf(sv[0][r] - nm);
      float p1 = __expf(sv[1][r] - nm);
      float ps = p0 + p1;
      #pragma unroll
      for (int off = 1; off < 16; off <<= 1) ps += __shfl_xor(ps, off, 32);
      lI[r] = lI[r] * alpha + ps;
      mI[r] = nm;
      #pragma unroll
      for (int t = 0; t < 4; ++t) oacc[t][r] = oacc[t][r] * alpha;
      sv[0][r] = p0; sv[1][r] = p1;
    }

    #pragma unroll
    for (int r = 0; r < 8; ++r) {
      ldsp[(r + hi8) * 40 + m15]      = f32_to_bf16(sv[0][r]);
      ldsp[(r + hi8) * 40 + m15 + 16] = f32_to_bf16(sv[1][r]);
    }
    __syncthreads();

    v16bf ap;
    {
      const unsigned short* base = ldsp + m15 * 40;
      ap = make_frag(*(const uint4*)(base + r0), *(const uint4*)(base + r0 + 16));
    }
    #pragma unroll
    for (int t = 0; t < 4; ++t) {
      const unsigned short* vb = vp + (long)kk * DK + t * 16 + m15;
      Bf16Frag fv;
      unsigned short* tv = (unsigned short*)&fv;
      #pragma unroll
      for (int s = 0; s < 8; ++s) { tv[s] = vb[(long)(r0 + s) * DK]; tv[s + 8] = vb[(long)(r0 + 16 + s) * DK]; }
      v16bf bv = make_frag(fv.lo, fv.hi);
      oacc[t] = wmma_bf16(ap, bv, oacc[t]);
    }
    __syncthreads();
  }

  #pragma unroll
  for (int t = 0; t < 4; ++t) {
    #pragma unroll
    for (int r = 0; r < 8; ++r) {
      int row = q0 + r + hi8;
      int col = h * DK + t * 16 + m15;
      o[((long)b * TT + row) * DD + col] = oacc[t][r] / lI[r];
    }
  }
}

// ---------------------------------------------------------------------------
// GLU: a * sigmoid(g) from [BT, 2D] -> [BT, D]
// ---------------------------------------------------------------------------
__global__ void glu_kernel(const float* __restrict__ hin, float* __restrict__ out) {
  long id = (long)blockIdx.x * 256 + threadIdx.x;  // < BT*D
  long row = id / DD;
  int c = (int)(id % DD);
  float a = hin[row * (2 * DD) + c];
  float g = hin[row * (2 * DD) + c + DD];
  out[id] = a * (1.f / (1.f + __expf(-g)));
}

// ---------------------------------------------------------------------------
// Depthwise conv (K=31, pad 15) + BatchNorm(eval) + Swish, out = bf16
// ---------------------------------------------------------------------------
__global__ void dwconv_bn_swish_kernel(
    const float* __restrict__ in, const float* __restrict__ w,
    const float* __restrict__ wb, const float* __restrict__ bg,
    const float* __restrict__ bb, const float* __restrict__ bm,
    const float* __restrict__ bv, unsigned short* __restrict__ outbf) {
  long id = (long)blockIdx.x * 256 + threadIdx.x;  // < BT*D
  int d = (int)(id % DD);
  long bt = id / DD;
  int t = (int)(bt % TT);
  long bT = bt - t;  // b*T
  float s = wb[d];
  #pragma unroll
  for (int k = 0; k < KW; ++k) {
    int tt = t + k - (KW / 2);
    if (tt >= 0 && tt < TT) s += in[(bT + tt) * DD + d] * w[d * KW + k];
  }
  s = (s - bm[d]) * rsqrtf(bv[d] + 1e-5f) * bg[d] + bb[d];
  s = s * (1.f / (1.f + __expf(-s)));  // swish
  outbf[id] = f32_to_bf16(s);
}

// ---------------------------------------------------------------------------
// Host-side orchestration
// ---------------------------------------------------------------------------
static inline size_t align256(size_t x) { return (x + 255) & ~size_t(255); }

extern "C" void kernel_launch(void* const* d_in, const int* in_sizes, int n_in,
                              void* d_out, int out_size, void* d_ws, size_t ws_size,
                              hipStream_t stream) {
  (void)in_sizes; (void)n_in; (void)out_size; (void)ws_size;
  const float* x          = (const float*)d_in[0];
  const unsigned char* mask = (const unsigned char*)d_in[1];
  const int* dialectid    = (const int*)d_in[2];
  const float* ln_ffmac_g = (const float*)d_in[3];
  const float* ln_ffmac_b = (const float*)d_in[4];
  const float* w1_mac     = (const float*)d_in[5];
  const float* b1_mac     = (const float*)d_in[6];
  const float* w2_mac     = (const float*)d_in[7];
  const float* b2_mac     = (const float*)d_in[8];
  const float* ln_mha_g   = (const float*)d_in[9];
  const float* ln_mha_b   = (const float*)d_in[10];
  const float* wq = (const float*)d_in[11]; const float* bq = (const float*)d_in[12];
  const float* wk = (const float*)d_in[13]; const float* bk = (const float*)d_in[14];
  const float* wv = (const float*)d_in[15]; const float* bv = (const float*)d_in[16];
  const float* wo = (const float*)d_in[17]; const float* bo = (const float*)d_in[18];
  const float* ln_conv_g  = (const float*)d_in[19];
  const float* ln_conv_b  = (const float*)d_in[20];
  const float* pw1_w = (const float*)d_in[21]; const float* pw1_b = (const float*)d_in[22];
  const float* dw_w  = (const float*)d_in[23]; const float* dw_b  = (const float*)d_in[24];
  const float* bn_g  = (const float*)d_in[25]; const float* bn_b  = (const float*)d_in[26];
  const float* bn_m  = (const float*)d_in[27]; const float* bn_v  = (const float*)d_in[28];
  const float* pw2_w = (const float*)d_in[29]; const float* pw2_b = (const float*)d_in[30];
  const float* ln_ff_g = (const float*)d_in[31];
  const float* ln_ff_b = (const float*)d_in[32];
  // d_in[33], d_in[34]: gate_w/gate_b — dead code (probs unused in output)
  const float* e_w1 = (const float*)d_in[35]; const float* e_b1 = (const float*)d_in[36];
  const float* e_w2 = (const float*)d_in[37]; const float* e_b2 = (const float*)d_in[38];
  const float* ln_final_g = (const float*)d_in[39];
  const float* ln_final_b = (const float*)d_in[40];
  float* out = (float*)d_out;

  // ---- workspace carve-up -------------------------------------------------
  char* base = (char*)d_ws;
  size_t cur = 0;
  auto takeBf = [&](long n) { unsigned short* p = (unsigned short*)(base + cur);
                              cur = align256(cur + (size_t)n * 2); return p; };
  auto takeF  = [&](long n) { float* p = (float*)(base + cur);
                              cur = align256(cur + (size_t)n * 4); return p; };
  auto takeI  = [&](long n) { int* p = (int*)(base + cur);
                              cur = align256(cur + (size_t)n * 4); return p; };

  unsigned short* w1_bf  = takeBf((long)DD * HH);
  unsigned short* w2_bf  = takeBf((long)HH * DD);
  unsigned short* wq_bf  = takeBf((long)DD * DD);
  unsigned short* wk_bf  = takeBf((long)DD * DD);
  unsigned short* wv_bf  = takeBf((long)DD * DD);
  unsigned short* wo_bf  = takeBf((long)DD * DD);
  unsigned short* pw1_bf = takeBf((long)DD * 2 * DD);
  unsigned short* pw2_bf = takeBf((long)DD * DD);
  unsigned short* ew1_bf = takeBf((long)EE * DD * HH);
  unsigned short* ew2_bf = takeBf((long)EE * HH * DD);
  int*            chosen = takeI(BB);
  float*          buf_x  = takeF((long)BT * DD);
  unsigned short* xn_bf  = takeBf((long)BT * DD);
  unsigned short* h_bf   = takeBf((long)BT * HH);
  float*          h_f32  = takeF((long)BT * HH);
  float*          q_f32  = takeF((long)BT * DD);   // also GLU-out / attn-out
  float*          k_f32  = takeF((long)BT * DD);
  float*          v_f32  = takeF((long)BT * DD);
  unsigned short* q_bf   = takeBf((long)BT * DD);
  unsigned short* k_bf   = takeBf((long)BT * DD);
  unsigned short* v_bf   = takeBf((long)BT * DD);

  auto cvt = [&](const float* src, unsigned short* dst, long n) {
    cvt_bf16_kernel<<<(unsigned)((n + 255) / 256), 256, 0, stream>>>(src, dst, n);
  };
  auto gemm = [&](const unsigned short* A, const unsigned short* Bw,
                  const float* bias, const float* res, float* Cf,
                  unsigned short* Cbf, const int* sel, int K, int N,
                  int rowsPerBatch, int nb, int mode, float rs) {
    dim3 g((unsigned)(N / 64), (unsigned)(rowsPerBatch / 128), (unsigned)nb);
    gemm_bf16_kernel<<<g, 256, 0, stream>>>(A, Bw, bias, res, Cf, Cbf, sel,
                                            K, N, rowsPerBatch, mode, rs);
  };
  auto ln = [&](const float* in, const float* g, const float* b,
                unsigned short* obf, float* of32) {
    ln_kernel<<<BT, 256, 0, stream>>>(in, g, b, obf, of32);
  };

  // ---- weight conversion (f32 -> bf16) ------------------------------------
  cvt(w1_mac, w1_bf, (long)DD * HH);
  cvt(w2_mac, w2_bf, (long)HH * DD);
  cvt(wq, wq_bf, (long)DD * DD);
  cvt(wk, wk_bf, (long)DD * DD);
  cvt(wv, wv_bf, (long)DD * DD);
  cvt(wo, wo_bf, (long)DD * DD);
  cvt(pw1_w, pw1_bf, (long)DD * 2 * DD);
  cvt(pw2_w, pw2_bf, (long)DD * DD);
  cvt(e_w1, ew1_bf, (long)EE * DD * HH);
  cvt(e_w2, ew2_bf, (long)EE * HH * DD);
  chosen_kernel<<<1, 32, 0, stream>>>(dialectid, chosen);

  (void)hipMemcpyAsync(buf_x, x, (size_t)BT * DD * 4, hipMemcpyDeviceToDevice,
                       stream);

  // ---- macaron FFN: x += 0.5 * (relu(LN(x) W1 + b1) W2 + b2) --------------
  ln(buf_x, ln_ffmac_g, ln_ffmac_b, xn_bf, nullptr);
  gemm(xn_bf, w1_bf, b1_mac, nullptr, nullptr, h_bf, nullptr, DD, HH, BT, 1, 1, 0.f);
  gemm(h_bf, w2_bf, b2_mac, buf_x, buf_x, nullptr, nullptr, HH, DD, BT, 1, 0, 0.5f);

  // ---- MHA ----------------------------------------------------------------
  ln(buf_x, ln_mha_g, ln_mha_b, xn_bf, nullptr);
  gemm(xn_bf, wq_bf, bq, nullptr, q_f32, nullptr, nullptr, DD, DD, BT, 1, 0, 0.f);
  gemm(xn_bf, wk_bf, bk, nullptr, k_f32, nullptr, nullptr, DD, DD, BT, 1, 0, 0.f);
  gemm(xn_bf, wv_bf, bv, nullptr, v_f32, nullptr, nullptr, DD, DD, BT, 1, 0, 0.f);
  pack_qkv_kernel<<<(long)BT * DD / 256, 256, 0, stream>>>(q_f32, k_f32, v_f32,
                                                           q_bf, k_bf, v_bf);
  {
    dim3 g(TT / 128, NHH, BB);
    flash_attn_kernel<<<g, 256, 0, stream>>>(q_bf, k_bf, v_bf, mask, q_f32);
  }
  cvt(q_f32, xn_bf, (long)BT * DD);  // attn output -> bf16
  gemm(xn_bf, wo_bf, bo, buf_x, buf_x, nullptr, nullptr, DD, DD, BT, 1, 0, 1.0f);

  // ---- conv module --------------------------------------------------------
  ln(buf_x, ln_conv_g, ln_conv_b, xn_bf, nullptr);
  gemm(xn_bf, pw1_bf, pw1_b, nullptr, h_f32, nullptr, nullptr, DD, 2 * DD, BT, 1, 0, 0.f);
  glu_kernel<<<(long)BT * DD / 256, 256, 0, stream>>>(h_f32, q_f32);
  dwconv_bn_swish_kernel<<<(long)BT * DD / 256, 256, 0, stream>>>(
      q_f32, dw_w, dw_b, bn_g, bn_b, bn_m, bn_v, xn_bf);
  gemm(xn_bf, pw2_bf, pw2_b, buf_x, buf_x, nullptr, nullptr, DD, DD, BT, 1, 0, 1.0f);

  // ---- MoE FFN (per-batch expert) ----------------------------------------
  ln(buf_x, ln_ff_g, ln_ff_b, xn_bf, nullptr);
  gemm(xn_bf, ew1_bf, e_b1, nullptr, nullptr, h_bf, chosen, DD, HH, TT, BB, 1, 0.f);
  gemm(h_bf, ew2_bf, e_b2, buf_x, buf_x, nullptr, chosen, HH, DD, TT, BB, 0, 0.5f);

  // ---- final LayerNorm -> d_out ------------------------------------------
  ln(buf_x, ln_final_g, ln_final_b, nullptr, out);
}
